// JIIF_18554258719328
// MI455X (gfx1250) — matrix operand order repaired and checked by reading, not compiled, not run on hardware
//
#include <hip/hip_runtime.h>
#include <hip/hip_bf16.h>

typedef __bf16 bf16;
typedef __attribute__((ext_vector_type(16))) __bf16 v16bf;
typedef __attribute__((ext_vector_type(8)))  __bf16 v8bf;
typedef __attribute__((ext_vector_type(8)))  float  v8f;

// ---- problem dims (fixed by the reference) ----
#define B_    2
#define CF_   128
#define HL_   64
#define WL_   64
#define HH_   256
#define WH_   256
#define NPT   (HH_*WH_)        // 65536 points per batch
#define MROWS (B_*NPT)         // 131072 total rows
#define MT    32               // rows (points) per block: 2 WMMA M-tiles
#define D0P   416              // 386 padded to mult of 32
#define D1    1024
#define D2    512
#define D3    256
#define D4    128
#define D5P   16               // 2 padded to 16

// ---- workspace layout (bf16 elements for weights, then f32 biases) ----
#define W1T_OFF 0
#define W1T_SZ  (D1*D0P)
#define W2T_OFF (W1T_OFF+W1T_SZ)
#define W2T_SZ  (D2*D1)
#define W3T_OFF (W2T_OFF+W2T_SZ)
#define W3T_SZ  (D3*D2)
#define W4T_OFF (W3T_OFF+W3T_SZ)
#define W4T_SZ  (D4*D3)
#define W5T_OFF (W4T_OFF+W4T_SZ)
#define W5T_SZ  (D5P*D4)
#define WT_TOTAL (W5T_OFF+W5T_SZ)
#define BIAS_BYTE_OFF ((size_t)((WT_TOTAL*2 + 255) & ~255))
#define B1_OFF 0
#define B2_OFF 1024
#define B3_OFF 1536
#define B4_OFF 1792
#define B5_OFF 1920

// Pack f32 weight [Ksrc][Nsrc] -> bf16 transposed [Npad][Kpad] (zeros pad).
__global__ void jiif_pack_wt(const float* __restrict__ src, bf16* __restrict__ dst,
                             int Ksrc, int Nsrc, int Kpad, int Npad) {
  int idx = blockIdx.x * blockDim.x + threadIdx.x;
  int tot = Kpad * Npad;
  if (idx >= tot) return;
  int o = idx / Kpad;
  int k = idx - o * Kpad;
  float v = (k < Ksrc && o < Nsrc) ? src[(size_t)k * Nsrc + o] : 0.f;
  dst[idx] = (bf16)v;
}

__global__ void jiif_pack_biases(const float* __restrict__ b1, const float* __restrict__ b2,
                                 const float* __restrict__ b3, const float* __restrict__ b4,
                                 const float* __restrict__ b5, float* __restrict__ dst) {
  int i = blockIdx.x * blockDim.x + threadIdx.x;
  if (i < 1024) dst[B1_OFF + i] = b1[i];
  if (i < 512)  dst[B2_OFF + i] = b2[i];
  if (i < 256)  dst[B3_OFF + i] = b3[i];
  if (i < 128)  dst[B4_OFF + i] = b4[i];
  if (i < 16)   dst[B5_OFF + i] = (i < 2) ? b5[i] : 0.f;
}

// One MLP layer: [32 x K] (LDS, bf16) x Wt[Nout][K] (global, bf16) -> [32 x Nout] (LDS, bf16)
// 8 waves split the Nout/16 column tiles. Each wave keeps TWO accumulators (two M-tiles)
// so every B-fragment fetched from L2 feeds two v_wmma_f32_16x16x32_bf16 ops.
__device__ __forceinline__ void mlp_layer(const bf16* __restrict__ inL, int K,
                                          const bf16* __restrict__ Wt,
                                          const float* __restrict__ bias,
                                          bf16* __restrict__ outL, int Nout, bool relu) {
  const int lane = threadIdx.x & 31;
  const int wave = threadIdx.x >> 5;
  const int nl   = lane & 15;
  const int half = lane >> 4;
  const bf16* arow0 = inL + nl * K + half * 8;          // M-tile 0: rows 0..15
  const bf16* arow1 = inL + (16 + nl) * K + half * 8;   // M-tile 1: rows 16..31
  const int ntiles = Nout >> 4;
  for (int t = wave; t < ntiles; t += 8) {
    const int n = (t << 4) + nl;
    const bf16* wrow = Wt + (size_t)n * K + half * 8;   // B: K 8h..8h+7 and 16+8h..16+8h+7 per lane
    v8f acc0 = {}, acc1 = {};
    for (int k0 = 0; k0 < K; k0 += 32) {
      union { v16bf v; v8bf h[2]; } a0, a1, b;
      b.h[0]  = *(const v8bf*)(wrow + k0);
      b.h[1]  = *(const v8bf*)(wrow + k0 + 16);
      a0.h[0] = *(const v8bf*)(arow0 + k0);
      a0.h[1] = *(const v8bf*)(arow0 + k0 + 16);
      a1.h[0] = *(const v8bf*)(arow1 + k0);
      a1.h[1] = *(const v8bf*)(arow1 + k0 + 16);
      acc0 = __builtin_amdgcn_wmma_f32_16x16x32_bf16(false, a0.v, false, b.v,
                                                     (short)0, acc0, false, false);
      acc1 = __builtin_amdgcn_wmma_f32_16x16x32_bf16(false, a1.v, false, b.v,
                                                     (short)0, acc1, false, false);
    }
    const float bn = bias[n];
#pragma unroll
    for (int v = 0; v < 8; ++v) {
      float x0 = acc0[v] + bn;
      float x1 = acc1[v] + bn;
      if (relu) { x0 = fmaxf(x0, 0.f); x1 = fmaxf(x1, 0.f); }
      outL[(size_t)(half * 8 + v) * Nout + n]      = (bf16)x0;
      outL[(size_t)(16 + half * 8 + v) * Nout + n] = (bf16)x1;
    }
  }
}

__global__ __launch_bounds__(256, 1)
void jiif_fused_kernel(const float* __restrict__ feat, const float* __restrict__ coord,
                       const float* __restrict__ hrg, const float* __restrict__ lrg,
                       const bf16* __restrict__ wt, const float* __restrict__ bias,
                       float* __restrict__ out) {
  __shared__ bf16  sA[MT * D1];          // act1 / act3      (64 KB)
  __shared__ bf16  sX[MT * D2];          // x(416)/act2/act4 (32 KB)
  __shared__ float sQgh[MT * CF_];       // q_guide_hr f32   (16 KB)
  __shared__ float sPred[4][MT][2];
  __shared__ int   sB[MT], sIy[MT], sIx[MT], sValid[MT];

  const int tid = threadIdx.x;
  const int r0  = blockIdx.x * MT;

  // ---- stage 0: HR-guide nearest indices + gather (reused by all 4 taps) ----
  if (tid < MT) {
    int r = r0 + tid;
    int b = r / NPT;
    int n = r - b * NPT;
    float cy = coord[((size_t)b * NPT + n) * 2 + 0];
    float cx = coord[((size_t)b * NPT + n) * 2 + 1];
    float fy = (cy + 1.f) * (HH_ * 0.5f) - 0.5f;
    float fx = (cx + 1.f) * (WH_ * 0.5f) - 0.5f;
    int iy = (int)floorf(fy + 0.5f);
    int ix = (int)floorf(fx + 0.5f);
    int valid = (iy >= 0) & (iy < HH_) & (ix >= 0) & (ix < WH_);
    iy = min(max(iy, 0), HH_ - 1);
    ix = min(max(ix, 0), WH_ - 1);
    sB[tid] = b; sIy[tid] = iy; sIx[tid] = ix; sValid[tid] = valid;
  }
  __syncthreads();
  for (int idx = tid; idx < MT * CF_; idx += 256) {
    int m = idx >> 7, c = idx & 127;
    float v = 0.f;
    if (sValid[m])
      v = hrg[(((size_t)sB[m] * CF_ + c) * HH_ + sIy[m]) * WH_ + sIx[m]];
    sQgh[m * CF_ + c] = v;
  }
  __syncthreads();

  // ---- 4 shifted taps ----
  for (int br = 0; br < 4; ++br) {
    const float vx = (br & 2) ? 1.f : -1.f;   // matches: for vx in (-1,1): for vy in (-1,1)
    const float vy = (br & 1) ? 1.f : -1.f;

    if (tid < MT) {
      int r = r0 + tid;
      int b = sB[tid];
      int n = r - b * NPT;
      float cy = coord[((size_t)b * NPT + n) * 2 + 0];
      float cx = coord[((size_t)b * NPT + n) * 2 + 1];
      float gy = cy + vx * (1.f / HL_);
      float gx = cx + vy * (1.f / WL_);
      float fy = (gy + 1.f) * (HL_ * 0.5f) - 0.5f;
      float fx = (gx + 1.f) * (WL_ * 0.5f) - 0.5f;
      int iy = (int)floorf(fy + 0.5f);
      int ix = (int)floorf(fx + 0.5f);
      int valid = (iy >= 0) & (iy < HL_) & (ix >= 0) & (ix < WL_);
      iy = min(max(iy, 0), HL_ - 1);
      ix = min(max(ix, 0), WL_ - 1);
      float qy = 0.f, qx = 0.f;
      if (valid) {
        qy = -1.f + (2.f * iy + 1.f) / HL_;
        qx = -1.f + (2.f * ix + 1.f) / WL_;
      }
      sIy[tid] = iy; sIx[tid] = ix; sValid[tid] = valid;
      sX[tid * D0P + 384] = (bf16)((cy - qy) * (float)HL_);
      sX[tid * D0P + 385] = (bf16)((cx - qx) * (float)WL_);
      for (int k = 386; k < D0P; ++k) sX[tid * D0P + k] = (bf16)0.f;
    }
    __syncthreads();

    for (int idx = tid; idx < MT * CF_; idx += 256) {
      int m = idx >> 7, c = idx & 127;
      float f = 0.f, l = 0.f;
      if (sValid[m]) {
        size_t base = (((size_t)sB[m] * CF_ + c) * HL_ + sIy[m]) * WL_ + sIx[m];
        f = feat[base];
        l = lrg[base];
      }
      float g = sQgh[m * CF_ + c];
      sX[m * D0P + c]       = (bf16)f;
      sX[m * D0P + 128 + c] = (bf16)g;
      sX[m * D0P + 256 + c] = (bf16)(g - l);
    }
    __syncthreads();

    mlp_layer(sX, D0P, wt + W1T_OFF, bias + B1_OFF, sA, D1, true);
    __syncthreads();
    mlp_layer(sA, D1, wt + W2T_OFF, bias + B2_OFF, sX, D2, true);
    __syncthreads();
    mlp_layer(sX, D2, wt + W3T_OFF, bias + B3_OFF, sA, D3, true);
    __syncthreads();
    mlp_layer(sA, D3, wt + W4T_OFF, bias + B4_OFF, sX, D4, true);
    __syncthreads();

    // ---- layer 5: one 16-col tile, two M-tiles, on wave 0 (cols 0,1 = value/logit) ----
    if ((tid >> 5) == 0) {
      const int lane = tid & 31;
      const int nl = lane & 15, half = lane >> 4;
      const bf16* arow0 = sX + nl * D4 + half * 8;
      const bf16* arow1 = sX + (16 + nl) * D4 + half * 8;
      const bf16* wrow  = wt + W5T_OFF + (size_t)nl * D4 + half * 8;
      v8f acc0 = {}, acc1 = {};
      for (int k0 = 0; k0 < D4; k0 += 32) {
        union { v16bf v; v8bf h[2]; } a0, a1, b;
        b.h[0]  = *(const v8bf*)(wrow + k0);
        b.h[1]  = *(const v8bf*)(wrow + k0 + 16);
        a0.h[0] = *(const v8bf*)(arow0 + k0);
        a0.h[1] = *(const v8bf*)(arow0 + k0 + 16);
        a1.h[0] = *(const v8bf*)(arow1 + k0);
        a1.h[1] = *(const v8bf*)(arow1 + k0 + 16);
        acc0 = __builtin_amdgcn_wmma_f32_16x16x32_bf16(false, a0.v, false, b.v,
                                                       (short)0, acc0, false, false);
        acc1 = __builtin_amdgcn_wmma_f32_16x16x32_bf16(false, a1.v, false, b.v,
                                                       (short)0, acc1, false, false);
      }
      if (nl < 2) {
        float bn = bias[B5_OFF + nl];
#pragma unroll
        for (int v = 0; v < 8; ++v) {
          sPred[br][half * 8 + v][nl]      = acc0[v] + bn;
          sPred[br][16 + half * 8 + v][nl] = acc1[v] + bn;
        }
      }
    }
    __syncthreads();
  }

  // ---- softmax over the 4 taps, weighted sum of values ----
  if (tid < MT) {
    float l0 = sPred[0][tid][1], l1 = sPred[1][tid][1];
    float l2 = sPred[2][tid][1], l3 = sPred[3][tid][1];
    float mx = fmaxf(fmaxf(l0, l1), fmaxf(l2, l3));
    float e0 = __expf(l0 - mx), e1 = __expf(l1 - mx);
    float e2 = __expf(l2 - mx), e3 = __expf(l3 - mx);
    float s = e0 + e1 + e2 + e3;
    float ret = (sPred[0][tid][0] * e0 + sPred[1][tid][0] * e1 +
                 sPred[2][tid][0] * e2 + sPred[3][tid][0] * e3) / s;
    out[r0 + tid] = ret;
  }
}

extern "C" void kernel_launch(void* const* d_in, const int* in_sizes, int n_in,
                              void* d_out, int out_size, void* d_ws, size_t ws_size,
                              hipStream_t stream) {
  const float* feat  = (const float*)d_in[0];
  const float* coord = (const float*)d_in[1];
  const float* hrg   = (const float*)d_in[2];
  const float* lrg   = (const float*)d_in[3];
  const float* w1 = (const float*)d_in[4];  const float* b1 = (const float*)d_in[5];
  const float* w2 = (const float*)d_in[6];  const float* b2 = (const float*)d_in[7];
  const float* w3 = (const float*)d_in[8];  const float* b3 = (const float*)d_in[9];
  const float* w4 = (const float*)d_in[10]; const float* b4 = (const float*)d_in[11];
  const float* w5 = (const float*)d_in[12]; const float* b5 = (const float*)d_in[13];

  bf16*  wt   = (bf16*)d_ws;
  float* bias = (float*)((char*)d_ws + BIAS_BYTE_OFF);

  // Pack weights (f32 [K][N] -> bf16 transposed [Npad][Kpad]) and biases.
  {
    int tot;
    tot = D0P * D1;  jiif_pack_wt<<<(tot + 255) / 256, 256, 0, stream>>>(w1, wt + W1T_OFF, 386, D1, D0P, D1);
    tot = D1  * D2;  jiif_pack_wt<<<(tot + 255) / 256, 256, 0, stream>>>(w2, wt + W2T_OFF, D1,  D2, D1,  D2);
    tot = D2  * D3;  jiif_pack_wt<<<(tot + 255) / 256, 256, 0, stream>>>(w3, wt + W3T_OFF, D2,  D3, D2,  D3);
    tot = D3  * D4;  jiif_pack_wt<<<(tot + 255) / 256, 256, 0, stream>>>(w4, wt + W4T_OFF, D3,  D4, D3,  D4);
    tot = D4  * D5P; jiif_pack_wt<<<(tot + 255) / 256, 256, 0, stream>>>(w5, wt + W5T_OFF, D4,  2,  D4,  D5P);
    jiif_pack_biases<<<4, 256, 0, stream>>>(b1, b2, b3, b4, b5, bias);
  }

  jiif_fused_kernel<<<MROWS / MT, 256, 0, stream>>>(feat, coord, hrg, lrg, wt, bias,
                                                    (float*)d_out);
}
